// yolo_v2_d19_62508954026344
// MI455X (gfx1250) — matrix operand order repaired and checked by reading, not compiled, not run on hardware
//
#include <hip/hip_runtime.h>
#include <hip/hip_bf16.h>
#include <math.h>

typedef __attribute__((ext_vector_type(16))) _Float16 v16h;
typedef __attribute__((ext_vector_type(8)))  _Float16 v8h;
typedef __attribute__((ext_vector_type(8)))  float    v8f;

#define HW19   361      // 19*19
#define NBOX   1805     // 361*5
#define NCLS   20
#define BOX_OFF   0
#define SCORE_OFF 7220
#define CLS_OFF   9025
#define KEEP_OFF  10830
#define ZPAGE_H   1280  // zero page: max C halves

// ---------------------------------------------------------------------------
__global__ void k_zero_f16(_Float16* __restrict__ p, int n) {
    int i = blockIdx.x * blockDim.x + threadIdx.x;
    if (i < n) p[i] = (_Float16)0.0f;
}

// ---------------------------------------------------------------------------
// f32 [C][HW] activation  ->  f16 [HW][C] (GEMM A layout, channel-contiguous)
// ---------------------------------------------------------------------------
__global__ void k_cvt_act(const float* __restrict__ in, _Float16* __restrict__ xt,
                          int C, int HW) {
    size_t idx = (size_t)blockIdx.x * blockDim.x + threadIdx.x;
    size_t total = (size_t)C * HW;
    if (idx >= total) return;
    int m  = (int)(idx / C);
    int ic = (int)(idx % C);
    xt[idx] = (_Float16)in[(size_t)ic * HW + m];
}

// ---------------------------------------------------------------------------
// f32 OIHW weight (oc, C, ks)  ->  f16 [oc][ks][C]  (B layout: K = ks*C + ic)
// Pads output channels [Nreal, Npad) with zeros.
// ---------------------------------------------------------------------------
__global__ void k_cvt_weight(const float* __restrict__ w, _Float16* __restrict__ wt,
                             int Npad, int Nreal, int C, int KS) {
    size_t idx = (size_t)blockIdx.x * blockDim.x + threadIdx.x;
    size_t total = (size_t)Npad * KS * C;
    if (idx >= total) return;
    int oc  = (int)(idx / ((size_t)KS * C));
    int rem = (int)(idx % ((size_t)KS * C));
    int ks  = rem / C;
    int ic  = rem % C;
    float v = (oc < Nreal) ? w[((size_t)oc * C + ic) * KS + ks] : 0.0f;
    wt[idx] = (_Float16)v;
}

// ---------------------------------------------------------------------------
// WMMA conv-as-GEMM.  A: f16 [HW][C] activations (implicit im2col for KS==9,
// pad=1).  Wt: f16 [N][KS*C].  One wave -> one 32(M)x16(N) output tile: two
// 16x16 accumulators share each B fragment (2 WMMAs per B load).  Padding /
// tail rows read from a zero page -> branch-free inner loop.
// Block = 256 threads (8 waves); grid = (ceil(M/32), Ntiles/8).
// ---------------------------------------------------------------------------
__global__ void __launch_bounds__(256)
k_wmma_gemm_conv(const _Float16* __restrict__ A, const _Float16* __restrict__ Wt,
                 const float* __restrict__ bias, const _Float16* __restrict__ zpage,
                 int C, int KS, int H, int W, int M, int N,
                 float* outF32, _Float16* outF16,
                 int out_stride, int out_col_off, int out_ncols, int do_leaky) {
    const int wave = threadIdx.x >> 5;
    const int lane = threadIdx.x & 31;
    const int l16  = lane & 15;
    const int hi   = lane >> 4;

    const int mtile = blockIdx.x;               // 32-row M tile
    const int ntile = blockIdx.y * 8 + wave;
    const int n     = ntile * 16 + l16;         // output column (oc) for this lane
    const int r0    = mtile * 32 + l16;         // A rows (spatial) for this lane
    const int r1    = r0 + 16;
    const int y0 = r0 / W, x0 = r0 % W;
    const int y1 = r1 / W, x1 = r1 % W;
    const int pad   = (KS == 9) ? 1 : 0;
    const size_t Ktot = (size_t)KS * C;

    // lane's B base: 16 contiguous halves at K = ks*C + icb + 16*hi
    const _Float16* wrow = Wt + (size_t)n * Ktot + 16 * hi;
    const int co0 = 16 * hi;       // A chunk offsets within a 32-wide K block
    const int co1 = 16 * hi + 8;

    v8f acc0 = {}, acc1 = {};
    for (int ks = 0; ks < KS; ++ks) {
        const int ky = ks / 3, kx = ks % 3;
        const int yy0 = y0 + ky - pad, xx0 = x0 + kx - pad;
        const int yy1 = y1 + ky - pad, xx1 = x1 + kx - pad;
        const bool v0 = (r0 < M) && (yy0 >= 0) && (yy0 < H) && (xx0 >= 0) && (xx0 < W);
        const bool v1 = (r1 < M) && (yy1 >= 0) && (yy1 < H) && (xx1 >= 0) && (xx1 < W);
        // invalid lanes read zeros from the zero page: branch-free inner loop
        const _Float16* arow0 = v0 ? (A + (size_t)(yy0 * W + xx0) * C) : zpage;
        const _Float16* arow1 = v1 ? (A + (size_t)(yy1 * W + xx1) * C) : zpage;
        const _Float16* brow  = wrow + (size_t)ks * C;

        for (int icb = 0; icb < C; icb += 32) {
            // A fragments (16x32 f16): hi=0 lanes hold K {0-15}, hi=1 {16-31}
            v8h a00 = *(const v8h*)(arow0 + icb + co0);
            v8h a01 = *(const v8h*)(arow0 + icb + co1);
            v8h a10 = *(const v8h*)(arow1 + icb + co0);
            v8h a11 = *(const v8h*)(arow1 + icb + co1);
            v16h av0, av1;
            #pragma unroll
            for (int q = 0; q < 8; ++q) {
                av0[q] = a00[q]; av0[8 + q] = a01[q];
                av1[q] = a10[q]; av1[8 + q] = a11[q];
            }
            // B fragment (32x16 f16): lane holds col n, K = icb + 16*hi + j
            v16h bv = *(const v16h*)(brow + icb);
            __builtin_prefetch((const void*)(brow + icb + 256), 0, 1);

            acc0 = __builtin_amdgcn_wmma_f32_16x16x32_f16(
                false, av0, false, bv, (short)0, acc0, false, false);
            acc1 = __builtin_amdgcn_wmma_f32_16x16x32_f16(
                false, av1, false, bv, (short)0, acc1, false, false);
        }
    }

    // Epilogue: D[row = base + v + 8*hi][col = n], two 16-row subtiles
    if (n < out_ncols) {
        const float b = bias[n];
        #pragma unroll
        for (int v = 0; v < 8; ++v) {
            const int rowA = mtile * 32 + v + 8 * hi;
            const int rowB = rowA + 16;
            if (rowA < M) {
                float val = acc0[v] + b;
                if (do_leaky) val = (val > 0.0f) ? val : 0.1f * val;
                const size_t o = (size_t)rowA * out_stride + out_col_off + n;
                if (outF16) outF16[o] = (_Float16)val;
                else        outF32[o] = val;
            }
            if (rowB < M) {
                float val = acc1[v] + b;
                if (do_leaky) val = (val > 0.0f) ? val : 0.1f * val;
                const size_t o = (size_t)rowB * out_stride + out_col_off + n;
                if (outF16) outF16[o] = (_Float16)val;
                else        outF32[o] = val;
            }
        }
    }
}

// ---------------------------------------------------------------------------
// 1x1 conv 512->64 on 38x38 + leaky + reorg(s=2) straight into xcat cols 0-255
// xcat[m][(i*2+j)*64 + c] = leaky(conv(f2)[c][2h+i][2w+j]),  m = h*19 + w
// ---------------------------------------------------------------------------
__global__ void k_convr_reorg(const float* __restrict__ f2, const float* __restrict__ wr,
                              const float* __restrict__ br, _Float16* __restrict__ xcat) {
    int t = blockIdx.x * blockDim.x + threadIdx.x;
    if (t >= HW19 * 256) return;
    int cc = t & 255;
    int m  = t >> 8;
    int c  = cc & 63;
    int ij = cc >> 6;
    int i  = ij >> 1, j = ij & 1;
    int h2 = m / 19, w2 = m % 19;
    int ys = 2 * h2 + i, xs = 2 * w2 + j;
    const float* src = f2 + ys * 38 + xs;
    const float* wv  = wr + (size_t)c * 512;
    float acc = br[c];
    for (int ic = 0; ic < 512; ++ic) acc += wv[ic] * src[(size_t)ic * 1444];
    acc = (acc > 0.0f) ? acc : 0.1f * acc;
    xcat[(size_t)m * 1280 + cc] = (_Float16)acc;
}

// ---------------------------------------------------------------------------
// Box decode: sigmoid/softmax/exp, anchors, clip; writes boxes/scores/cls,
// zero-inits keep flags.
// ---------------------------------------------------------------------------
__global__ void k_decode(const float* __restrict__ pred, float* __restrict__ out) {
    const float AW[5] = {1.19f, 2.79f, 4.53f, 8.06f, 10.32f};
    const float AH[5] = {1.98f, 4.59f, 8.92f, 5.29f, 10.65f};
    int i = blockIdx.x * blockDim.x + threadIdx.x;
    if (i >= NBOX) return;
    int m = i / 5, a = i % 5;
    int gx = m % 19, gy = m / 19;
    const float* row = pred + (size_t)m * 125;

    float conf = row[a];
    float ml = -1e30f; int mi = 0;
    #pragma unroll
    for (int c = 0; c < NCLS; ++c) {
        float l = row[5 + a * NCLS + c];
        if (l > ml) { ml = l; mi = c; }
    }
    float den = 0.0f;
    #pragma unroll
    for (int c = 0; c < NCLS; ++c) den += expf(row[5 + a * NCLS + c] - ml);
    float score = (1.0f / (1.0f + expf(-conf))) * (1.0f / den);

    float tx = row[105 + a * 4 + 0], ty = row[105 + a * 4 + 1];
    float tw = row[105 + a * 4 + 2], th = row[105 + a * 4 + 3];
    float cx = (1.0f / (1.0f + expf(-tx)) + (float)gx) * 32.0f;
    float cy = (1.0f / (1.0f + expf(-ty)) + (float)gy) * 32.0f;
    float bw = expf(tw) * AW[a] * 32.0f;
    float bh = expf(th) * AH[a] * 32.0f;
    float x1 = fminf(fmaxf((cx - 0.5f * bw) * (1.0f / 608.0f), 0.0f), 1.0f);
    float y1 = fminf(fmaxf((cy - 0.5f * bh) * (1.0f / 608.0f), 0.0f), 1.0f);
    float x2 = fminf(fmaxf((cx + 0.5f * bw) * (1.0f / 608.0f), 0.0f), 1.0f);
    float y2 = fminf(fmaxf((cy + 0.5f * bh) * (1.0f / 608.0f), 0.0f), 1.0f);

    out[BOX_OFF + i * 4 + 0] = x1;
    out[BOX_OFF + i * 4 + 1] = y1;
    out[BOX_OFF + i * 4 + 2] = x2;
    out[BOX_OFF + i * 4 + 3] = y2;
    out[SCORE_OFF + i] = score;
    out[CLS_OFF + i]   = (float)mi;
    out[KEEP_OFF + i]  = 0.0f;
}

// ---------------------------------------------------------------------------
// Greedy per-class NMS.  One block per class; class masks partition boxes so
// blocks write disjoint keep flags.  Suppression state lives in LDS.
// ---------------------------------------------------------------------------
__global__ void __launch_bounds__(256) k_nms(float* __restrict__ out) {
    const int cls_id = blockIdx.x;
    const int tid = threadIdx.x;
    const float* boxes  = out + BOX_OFF;
    const float* scores = out + SCORE_OFF;
    const float* cls    = out + CLS_OFF;
    float* keep         = out + KEEP_OFF;

    __shared__ unsigned char sup[NBOX];
    __shared__ float sVal[256];
    __shared__ int   sIdx[256];
    __shared__ float bb[5];   // x1,y1,x2,y2,area of current best
    __shared__ int   bestIdx;

    for (int i = tid; i < NBOX; i += 256) sup[i] = 0;
    __syncthreads();

    for (int iter = 0; iter < NBOX; ++iter) {
        // argmax over un-suppressed boxes of this class (ties -> lower index)
        float best = -1e30f; int bi = -1;
        for (int i = tid; i < NBOX; i += 256) {
            if (cls[i] == (float)cls_id && !sup[i]) {
                float s = scores[i];
                if (s > best || (s == best && (bi < 0 || i < bi))) { best = s; bi = i; }
            }
        }
        sVal[tid] = best; sIdx[tid] = bi;
        __syncthreads();
        for (int off = 128; off > 0; off >>= 1) {
            if (tid < off) {
                float v2 = sVal[tid + off]; int i2 = sIdx[tid + off];
                if (i2 >= 0 && (v2 > sVal[tid] ||
                    (v2 == sVal[tid] && (sIdx[tid] < 0 || i2 < sIdx[tid])))) {
                    sVal[tid] = v2; sIdx[tid] = i2;
                }
            }
            __syncthreads();
        }
        if (tid == 0) {
            int b = sIdx[0];
            bestIdx = b;
            if (b >= 0) {
                sup[b] = 1;
                keep[b] = 1.0f;
                float x1 = boxes[b * 4 + 0], y1 = boxes[b * 4 + 1];
                float x2 = boxes[b * 4 + 2], y2 = boxes[b * 4 + 3];
                bb[0] = x1; bb[1] = y1; bb[2] = x2; bb[3] = y2;
                bb[4] = (x2 - x1) * (y2 - y1);
            }
        }
        __syncthreads();
        int b = bestIdx;
        if (b < 0) break;
        // suppress same-class boxes with IoU > 0.5 against best
        for (int i = tid; i < NBOX; i += 256) {
            if (cls[i] == (float)cls_id && !sup[i]) {
                float x1 = boxes[i * 4 + 0], y1 = boxes[i * 4 + 1];
                float x2 = boxes[i * 4 + 2], y2 = boxes[i * 4 + 3];
                float ar = (x2 - x1) * (y2 - y1);
                float iw = fmaxf(1e-10f, fminf(x2, bb[2]) - fmaxf(x1, bb[0]));
                float ih = fmaxf(1e-10f, fminf(y2, bb[3]) - fmaxf(y1, bb[1]));
                float inter = iw * ih;
                float iou = inter / (ar + bb[4] - inter);
                if (iou > 0.5f) sup[i] = 1;
            }
        }
        __syncthreads();
    }
}

// ---------------------------------------------------------------------------
extern "C" void kernel_launch(void* const* d_in, const int* in_sizes, int n_in,
                              void* d_out, int out_size, void* d_ws, size_t ws_size,
                              hipStream_t stream) {
    (void)in_sizes; (void)n_in; (void)out_size; (void)ws_size;
    const float* f2  = (const float*)d_in[0];   // (1,512,38,38)
    const float* f3  = (const float*)d_in[1];   // (1,1024,19,19)
    const float* w1a = (const float*)d_in[2];
    const float* b1a = (const float*)d_in[3];
    const float* w1b = (const float*)d_in[4];
    const float* b1b = (const float*)d_in[5];
    const float* wr  = (const float*)d_in[6];
    const float* br  = (const float*)d_in[7];
    const float* w2  = (const float*)d_in[8];
    const float* b2  = (const float*)d_in[9];
    const float* wp  = (const float*)d_in[10];
    const float* bp  = (const float*)d_in[11];
    float* out = (float*)d_out;

    // workspace carve-up (weight scratch reused across convs)
    char* p = (char*)d_ws;
    auto carve = [&](size_t bytes) { void* r = (void*)p; p += (bytes + 255) & ~(size_t)255; return r; };
    _Float16* x_t3  = (_Float16*)carve((size_t)HW19 * 1024 * 2);
    _Float16* x_ta  = (_Float16*)carve((size_t)HW19 * 1024 * 2);
    _Float16* xcat  = (_Float16*)carve((size_t)HW19 * 1280 * 2);
    _Float16* x_tc  = (_Float16*)carve((size_t)HW19 * 1024 * 2);
    float*    pred  = (float*)   carve((size_t)HW19 * 125 * 4);
    _Float16* zpage = (_Float16*)carve((size_t)ZPAGE_H * 2);
    _Float16* wtbuf = (_Float16*)carve((size_t)1024 * 9 * 1280 * 2);  // max weight

    const int T = 256;
    auto blocks = [](size_t n) { return (unsigned)((n + 255) / 256); };

    // 0) zero page for padded/tail A rows
    k_zero_f16<<<blocks(ZPAGE_H), T, 0, stream>>>(zpage, ZPAGE_H);

    // 1) feat3 -> f16 [361][1024]
    k_cvt_act<<<blocks((size_t)1024 * HW19), T, 0, stream>>>(f3, x_t3, 1024, HW19);

    // 2) conv1a (3x3, 1024->1024) -> leaky f16 x_ta
    k_cvt_weight<<<blocks((size_t)1024 * 9 * 1024), T, 0, stream>>>(w1a, wtbuf, 1024, 1024, 1024, 9);
    k_wmma_gemm_conv<<<dim3(12, 8), T, 0, stream>>>(x_t3, wtbuf, b1a, zpage,
        1024, 9, 19, 19, HW19, 1024, nullptr, x_ta, 1024, 0, 1024, 1);

    // 3) conv1b (3x3, 1024->1024) -> leaky f16 into xcat cols 256..1279
    k_cvt_weight<<<blocks((size_t)1024 * 9 * 1024), T, 0, stream>>>(w1b, wtbuf, 1024, 1024, 1024, 9);
    k_wmma_gemm_conv<<<dim3(12, 8), T, 0, stream>>>(x_ta, wtbuf, b1b, zpage,
        1024, 9, 19, 19, HW19, 1024, nullptr, xcat, 1280, 256, 1024, 1);

    // 4) reorg branch (1x1 512->64 on 38x38, leaky, reorg) -> xcat cols 0..255
    k_convr_reorg<<<blocks((size_t)HW19 * 256), T, 0, stream>>>(f2, wr, br, xcat);

    // 5) conv2 (3x3, 1280->1024) -> leaky f16 x_tc
    k_cvt_weight<<<blocks((size_t)1024 * 9 * 1280), T, 0, stream>>>(w2, wtbuf, 1024, 1024, 1280, 9);
    k_wmma_gemm_conv<<<dim3(12, 8), T, 0, stream>>>(xcat, wtbuf, b2, zpage,
        1280, 9, 19, 19, HW19, 1024, nullptr, x_tc, 1024, 0, 1024, 1);

    // 6) predictor (1x1, 1024->125, N padded to 128) -> f32 pred, no activation
    k_cvt_weight<<<blocks((size_t)128 * 1024), T, 0, stream>>>(wp, wtbuf, 128, 125, 1024, 1);
    k_wmma_gemm_conv<<<dim3(12, 1), T, 0, stream>>>(x_tc, wtbuf, bp, zpage,
        1024, 1, 19, 19, HW19, 128, pred, nullptr, 125, 0, 125, 0);

    // 7) decode boxes/scores/cls + zero keep
    k_decode<<<blocks(NBOX), T, 0, stream>>>(pred, out);

    // 8) per-class greedy NMS (20 blocks)
    k_nms<<<NCLS, T, 0, stream>>>(out);
}
